// PBRNN_90056874262925
// MI455X (gfx1250) — matrix operand order repaired and checked by reading, not compiled, not run on hardware
//
#include <hip/hip_runtime.h>
#include <hip/hip_bf16.h>

// ---------------------------------------------------------------------------
// PBRNN (single-layer LSTM) for MI455X / gfx1250, wave32, WMMA bf16.
//   Xg = X @ W_ih^T + (b_ih + b_hh)   (parallel GEMM, chunked over sequence)
//   per step: gates = Xg[t] + h @ W_hh^T ; LSTM update (torch gate order ifgo)
// Serial path (k_cell): double-buffered async global->LDS staging of h and
// the block's W_hh slice, overlapped with v_wmma_f32_16x16x32_bf16.
// ---------------------------------------------------------------------------

typedef __attribute__((ext_vector_type(16))) __bf16 v16bf;
typedef __attribute__((ext_vector_type(8)))  __bf16 v8bf;
typedef __attribute__((ext_vector_type(8)))  float  v8f;
typedef __attribute__((ext_vector_type(4)))  int    v4i;

#define BATCH   128
#define SEQ     512
#define HID     1024
#define GATES   4096   // 4*HID
#define KDIM    1024   // both W_ih and W_hh have K=1024
#define CH      32     // sequence chunk for precomputed Xg
#define NCHUNK  (SEQ / CH)

#if __has_builtin(__builtin_amdgcn_global_load_async_to_lds_b128) && \
    __has_builtin(__builtin_amdgcn_s_wait_asynccnt)
#define USE_ASYNC 1
#else
#define USE_ASYNC 0
#endif

static __device__ __forceinline__ v16bf cat16(v8bf lo, v8bf hi) {
  return __builtin_shufflevector(lo, hi, 0,1,2,3,4,5,6,7,8,9,10,11,12,13,14,15);
}
static __device__ __forceinline__ float sigmoidf_(float x) {
  return 1.0f / (1.0f + __expf(-x));
}

#if USE_ASYNC
static __device__ __forceinline__ void async_cp16(const void* g, void* l) {
  __builtin_amdgcn_global_load_async_to_lds_b128(
      (__attribute__((address_space(1))) v4i*)g,
      (__attribute__((address_space(3))) v4i*)l, 0, 0);
}
#endif
static __device__ __forceinline__ void sync_cp16(const void* g, void* l) {
  *(uint4*)l = *(const uint4*)g;
}

// ---------------------------------------------------------------------------
// Weight swizzle: WMMA B operand (32x16 bf16 tile) fragment layout:
//   frag f = (n_tile, k_chunk); element (lane, half):
//     n = n_tile*16 + (lane & 15)
//     k = k_chunk*32 + (lane>=16 ? 16 : 0) + half
//   flat = ((n_tile*(K/32) + k_chunk)*32 + lane)*16 + half
// ---------------------------------------------------------------------------
__global__ __launch_bounds__(256) void k_prep_weights(
    const float* __restrict__ Wih, const float* __restrict__ Whh,
    __bf16* __restrict__ fih, __bf16* __restrict__ fhh) {
  int tid = blockIdx.x * 256 + threadIdx.x;           // 0 .. 2*4096*1024-1
  int which = tid >= GATES * KDIM;
  int t = which ? tid - GATES * KDIM : tid;
  int e    = t & 511;
  int fid  = t >> 9;
  int lane = e >> 4;
  int half = e & 15;
  int kc   = fid & 31;
  int nt   = fid >> 5;
  int n = nt * 16 + (lane & 15);
  int k = kc * 32 + ((lane >> 4) << 4) + half;
  const float* src = which ? Whh : Wih;
  __bf16*      dst = which ? fhh : fih;
  dst[t] = (__bf16)src[n * KDIM + k];
}

__global__ __launch_bounds__(256) void k_init(
    const float* __restrict__ bih, const float* __restrict__ bhh,
    float* __restrict__ bias, __bf16* __restrict__ h0,
    __bf16* __restrict__ h1, float* __restrict__ c) {
  int tid = blockIdx.x * 256 + threadIdx.x;
  if (tid < GATES) bias[tid] = bih[tid] + bhh[tid];
  h0[tid] = (__bf16)0.0f;
  h1[tid] = (__bf16)0.0f;
  c[tid]  = 0.0f;
}

// ---------------------------------------------------------------------------
// Input projection for one sequence chunk (off the critical path):
//   Xg[tc][b][n] = sum_k X[b][t0+tc][k] * Wih[n][k] + bias[n]
// grid = (CH, GATES/64).  Block tile: 128 batch rows x 64 gate cols.
// W sub-slice staged in LDS per K-chunk (kills 8x inter-wave refetch).
// ---------------------------------------------------------------------------
__global__ __launch_bounds__(256) void k_xproj(
    const float* __restrict__ X, const __bf16* __restrict__ WF,
    const float* __restrict__ bias, float* __restrict__ Xg, int t0) {
  __shared__ __align__(16) __bf16 zt[128][136];   // 128x128 bf16, +16B row pad
  __shared__ __align__(16) __bf16 wx[8192];       // 4 ntiles x 4 kchunks x 1KB
  const int tc   = blockIdx.x;
  const int nb   = blockIdx.y;
  const int tid  = threadIdx.x;
  const int wave = tid >> 5;
  const int lane = tid & 31;
  const int n0   = nb * 64;

  v8f acc[4] = {};

  for (int kb = 0; kb < KDIM / 128; ++kb) {
    __syncthreads();
    // stage X tile [128 x 128] f32 -> bf16, coalesced float4 loads
    #pragma unroll
    for (int i = 0; i < 16; ++i) {
      int f  = tid + i * 256;
      int r  = f >> 5;
      int c4 = f & 31;
      const float4 v = *(const float4*)(X + ((size_t)r * SEQ + (t0 + tc)) * KDIM
                                          + kb * 128 + c4 * 4);
      __bf16* d = &zt[r][c4 * 4];
      d[0] = (__bf16)v.x; d[1] = (__bf16)v.y;
      d[2] = (__bf16)v.z; d[3] = (__bf16)v.w;
    }
    // stage W sub-slice [4 ntiles x 4 kchunks] into LDS (16 KB)
    #pragma unroll
    for (int i = 0; i < 4; ++i) {
      int f    = tid + i * 256;        // 1024 uint4s
      int nt   = f >> 8;
      int r2   = f & 255;
      int kc2  = r2 >> 6;
      int q    = r2 & 63;
      int lsrc = q >> 1;
      int u    = q & 1;
      size_t frag = (size_t)(((n0 >> 4) + nt) * 32 + kb * 4 + kc2);
      sync_cp16(WF + frag * 512 + lsrc * 16 + u * 8,
                &wx[(((nt * 4 + kc2) * 2 + u) * 32 + lsrc) * 8]);
    }
    __syncthreads();

    const int aRow = wave * 16 + (lane & 15);
    const int kSel = (lane >= 16) ? 16 : 0;
    #pragma unroll
    for (int kc2 = 0; kc2 < 4; ++kc2) {
      const int kk = kc2 * 32;
      v16bf a = cat16(*(const v8bf*)&zt[aRow][kk + kSel],
                      *(const v8bf*)&zt[aRow][kk + kSel + 8]);
      #pragma unroll
      for (int g = 0; g < 4; ++g) {
        const __bf16* bp = &wx[((g * 4 + kc2) * 2) * 256 + lane * 8];
        v16bf b = cat16(*(const v8bf*)bp, *(const v8bf*)(bp + 256));
        acc[g] = __builtin_amdgcn_wmma_f32_16x16x32_bf16(
            false, a, false, b, (short)0, acc[g], false, false);
      }
    }
  }

  const int colL = lane & 15;
  const int rOff = (lane >= 16) ? 8 : 0;
  #pragma unroll
  for (int g = 0; g < 4; ++g) {
    const int n  = n0 + g * 16 + colL;
    const float bv = bias[n];
    #pragma unroll
    for (int r = 0; r < 8; ++r) {
      const int b = wave * 16 + r + rOff;
      Xg[((size_t)tc * BATCH + b) * GATES + n] = acc[g][r] + bv;
    }
  }
}

// ---------------------------------------------------------------------------
// One LSTM step.  grid = 64 blocks; block blk owns hidden units [16*blk,+16).
// N-tiles g=0..3 are gate columns {j, H+j, 2H+j, 3H+j}: after the GEMM each
// (lane, acc-row) holds i,f,g,o for one (batch, hidden) element.
// K pipelined in 16 stages of 64, double-buffered h+W tiles in LDS,
// staged via async global->LDS when available.  h ping-pongs across steps.
// ---------------------------------------------------------------------------
__global__ __launch_bounds__(256) void k_cell(
    const float* __restrict__ Xg_t, const __bf16* __restrict__ WF,
    const __bf16* __restrict__ hin, __bf16* __restrict__ hout,
    float* __restrict__ cst, float* __restrict__ out, int last) {
  __shared__ __align__(16) __bf16 hbuf[2][128][72];  // 128x64 bf16 +16B row pad
  __shared__ __align__(16) __bf16 wbuf[2][4096];     // 4 ntiles x 2 kchunks x 1KB
  const int blk  = blockIdx.x;
  const int tid  = threadIdx.x;
  const int wave = tid >> 5;
  const int lane = tid & 31;

  v8f acc[4] = {};

  // per-stage staging: 4 h-tile copies + 2 W copies per thread (16B each)
  auto stage_addrs = [&](int s, int i, const void** g, void** l, int part) {
    if (part == 0) {                       // h tile: 1024 uint4s
      int f = tid + i * 256;
      int r = f >> 3, c = f & 7;
      *g = hin + (size_t)r * HID + s * 64 + c * 8;
      *l = &hbuf[s & 1][r][c * 8];
    } else {                               // W slice: 512 uint4s
      int f    = tid + i * 256;
      int nt   = f >> 7;
      int r2   = f & 127;
      int kc2  = (r2 >> 6) & 1;
      int q    = r2 & 63;
      int lsrc = q >> 1;
      int u    = q & 1;
      size_t frag = (size_t)((nt * 64 + blk) * 32 + (s * 2 + kc2));
      *g = WF + frag * 512 + lsrc * 16 + u * 8;
      *l = &wbuf[s & 1][(((nt * 2 + kc2) * 2 + u) * 32 + lsrc) * 8];
    }
  };
#if USE_ASYNC
  auto issue_stage = [&](int s) {
    const void* g; void* l;
    #pragma unroll
    for (int i = 0; i < 4; ++i) { stage_addrs(s, i, &g, &l, 0); async_cp16(g, l); }
    #pragma unroll
    for (int i = 0; i < 2; ++i) { stage_addrs(s, i, &g, &l, 1); async_cp16(g, l); }
  };
#else
  auto issue_stage = [&](int s) {
    const void* g; void* l;
    #pragma unroll
    for (int i = 0; i < 4; ++i) { stage_addrs(s, i, &g, &l, 0); sync_cp16(g, l); }
    #pragma unroll
    for (int i = 0; i < 2; ++i) { stage_addrs(s, i, &g, &l, 1); sync_cp16(g, l); }
  };
#endif
  auto compute_stage = [&](int s) {
    const int buf  = s & 1;
    const int aRow = wave * 16 + (lane & 15);
    const int kSel = (lane >= 16) ? 16 : 0;
    #pragma unroll
    for (int kc2 = 0; kc2 < 2; ++kc2) {
      const int kk = kc2 * 32;
      v16bf a = cat16(*(const v8bf*)&hbuf[buf][aRow][kk + kSel],
                      *(const v8bf*)&hbuf[buf][aRow][kk + kSel + 8]);
      #pragma unroll
      for (int g = 0; g < 4; ++g) {
        const __bf16* bp = &wbuf[buf][((g * 2 + kc2) * 2) * 256 + lane * 8];
        v16bf b = cat16(*(const v8bf*)bp, *(const v8bf*)(bp + 256));
        acc[g] = __builtin_amdgcn_wmma_f32_16x16x32_bf16(
            false, a, false, b, (short)0, acc[g], false, false);
      }
    }
  };

#if USE_ASYNC
  issue_stage(0);
  for (int s = 0; s < 16; ++s) {
    if (s + 1 < 16) {
      issue_stage(s + 1);
      __builtin_amdgcn_s_wait_asynccnt(6);   // stage s complete, s+1 in flight
    } else {
      __builtin_amdgcn_s_wait_asynccnt(0);
    }
    __syncthreads();          // all waves' stage-s data visible
    compute_stage(s);
    __syncthreads();          // everyone done reading before buffer reuse
  }
#else
  for (int s = 0; s < 16; ++s) {
    __syncthreads();
    issue_stage(s);
    __syncthreads();
    compute_stage(s);
  }
#endif

  // elementwise LSTM update: fully register-local per (lane, r)
  const int j    = blk * 16 + (lane & 15);
  const int rOff = (lane >= 16) ? 8 : 0;
  #pragma unroll
  for (int r = 0; r < 8; ++r) {
    const int b = wave * 16 + r + rOff;
    const float xi = Xg_t[(size_t)b * GATES + j];
    const float xf = Xg_t[(size_t)b * GATES + HID + j];
    const float xg = Xg_t[(size_t)b * GATES + 2 * HID + j];
    const float xo = Xg_t[(size_t)b * GATES + 3 * HID + j];
    const float iv = sigmoidf_(acc[0][r] + xi);
    const float fv = sigmoidf_(acc[1][r] + xf);
    const float gv = tanhf(acc[2][r] + xg);
    const float ov = sigmoidf_(acc[3][r] + xo);
    const int idx = b * HID + j;
    const float cn = fv * cst[idx] + iv * gv;
    cst[idx] = cn;
    const float hn = ov * tanhf(cn);
    hout[idx] = (__bf16)hn;
    if (last) {                 // d_out = stack([h, c]) as f32
      out[idx]              = hn;
      out[BATCH * HID + idx] = cn;
    }
  }
}

// ---------------------------------------------------------------------------
extern "C" void kernel_launch(void* const* d_in, const int* in_sizes, int n_in,
                              void* d_out, int out_size, void* d_ws, size_t ws_size,
                              hipStream_t stream) {
  const float* X   = (const float*)d_in[0];
  const float* Wih = (const float*)d_in[1];
  const float* Whh = (const float*)d_in[2];
  const float* bih = (const float*)d_in[3];
  const float* bhh = (const float*)d_in[4];

  size_t off = 0;
  auto alloc = [&](size_t bytes) -> void* {
    void* p = (char*)d_ws + off;
    off += (bytes + 255) & ~(size_t)255;
    return p;
  };
  __bf16* wf_ih = (__bf16*)alloc((size_t)GATES * KDIM * 2);   // 8 MB
  __bf16* wf_hh = (__bf16*)alloc((size_t)GATES * KDIM * 2);   // 8 MB
  float*  bias  = (float*) alloc((size_t)GATES * 4);
  __bf16* h0    = (__bf16*)alloc((size_t)BATCH * HID * 2);
  __bf16* h1    = (__bf16*)alloc((size_t)BATCH * HID * 2);
  float*  cbuf  = (float*) alloc((size_t)BATCH * HID * 4);
  float*  xg    = (float*) alloc((size_t)CH * BATCH * GATES * 4); // 64 MB
  (void)ws_size; (void)in_sizes; (void)n_in; (void)out_size;

  k_prep_weights<<<(2 * GATES * KDIM) / 256, 256, 0, stream>>>(Wih, Whh, wf_ih, wf_hh);
  k_init<<<(BATCH * HID) / 256, 256, 0, stream>>>(bih, bhh, bias, h0, h1, cbuf);

  for (int ch = 0; ch < NCHUNK; ++ch) {
    k_xproj<<<dim3(CH, GATES / 64), 256, 0, stream>>>(X, wf_ih, bias, xg, ch * CH);
    for (int tc = 0; tc < CH; ++tc) {
      const int t = ch * CH + tc;
      __bf16* hin  = (t & 1) ? h1 : h0;
      __bf16* hout = (t & 1) ? h0 : h1;
      k_cell<<<HID / 16, 256, 0, stream>>>(xg + (size_t)tc * BATCH * GATES,
                                           wf_hh, hin, hout, cbuf,
                                           (float*)d_out, t == SEQ - 1);
    }
  }
}
